// MultiHeadSelfAttention_79757542687093
// MI455X (gfx1250) — compile-verified, matrix operand order checked by reading, not run on hardware
//
#include <hip/hip_runtime.h>

// ---------------- problem constants (from reference) ----------------
#define C_   256
#define NH_  8
#define HD_  32      // C/NH
#define N_   2304    // 48*48
#define SCALE_Q 0.17677669529663687f   // 1/sqrt(32)

typedef __attribute__((ext_vector_type(16))) _Float16 v16h;
typedef __attribute__((ext_vector_type(8)))  _Float16 v8h;
typedef __attribute__((ext_vector_type(8)))  float    v8f;

__device__ __forceinline__ v8f wmma_f16(v16h a, v16h b, v8f c) {
  // D(f32 16x16) = A(f16 16x32) * B(f16 32x16) + C
  return __builtin_amdgcn_wmma_f32_16x16x32_f16(false, a, false, b,
                                                (short)0, c, false, false);
}

__device__ __forceinline__ v8h ld8(const _Float16* p) {
  return *(const v8h*)p;   // 16-byte load -> global_load_b128
}

__device__ __forceinline__ v16h cat8(v8h a, v8h b) {
  return __builtin_shufflevector(a, b, 0,1,2,3,4,5,6,7,8,9,10,11,12,13,14,15);
}

// =====================================================================
// Kernel 1: fused QKV projection (1x1 conv == GEMM over channels)
//   y[o,n] = sum_c W[o,c] x[c,n] + bias[o]     (per batch, per matrix)
// Outputs (f16 workspace):
//   qws[b,h,n,hd]  (Q^T per head, pre-scaled by 1/sqrt(hd))
//   kws[b,h,n,hd]  (K^T per head)
//   vws[b,h,hd,n]  (V per head, d-major)
// grid = (N/16, 4, B*3), block = 128 (4 waves); wave -> 16x16 output tile
// =====================================================================
__global__ __launch_bounds__(128) void qkv_proj_kernel(
    const float* __restrict__ x,
    const float* __restrict__ wq, const float* __restrict__ bq,
    const float* __restrict__ wk, const float* __restrict__ bk,
    const float* __restrict__ wv, const float* __restrict__ bv,
    _Float16* __restrict__ qws, _Float16* __restrict__ kws,
    _Float16* __restrict__ vws)
{
  const int lane = threadIdx.x & 31;
  const int wid  = threadIdx.x >> 5;
  const int l2   = lane & 15;
  const int hi   = lane >> 4;

  const int mat = blockIdx.z % 3;
  const int b   = blockIdx.z / 3;
  const float* W    = (mat == 0) ? wq : (mat == 1) ? wk : wv;
  const float* bias = (mat == 0) ? bq : (mat == 1) ? bk : bv;
  const float* X = x + (size_t)b * C_ * N_;

  const int obase = (blockIdx.y * 4 + wid) * 16;  // 0..240
  const int n     = blockIdx.x * 16 + l2;

  v8f acc = {0.f,0.f,0.f,0.f,0.f,0.f,0.f,0.f};

  for (int kk = 0; kk < C_; kk += 32) {
    // A operand: W rows (16 o) x 32 c.  lane: o = obase+l2,
    //   halves 0..7 -> c = kk+hi*8+i, halves 8..15 -> c = kk+16+hi*8+i
    const float* wr = W + (size_t)(obase + l2) * C_ + kk + hi * 8;
    v16h a, bb;
#pragma unroll
    for (int i = 0; i < 8; ++i) {
      a[i]     = (_Float16)wr[i];
      a[8 + i] = (_Float16)wr[16 + i];
    }
    // B operand: X block (32 c) x (16 n).  lane: col n,
    //   halves 0..7 -> c = kk+hi*8+i, halves 8..15 -> c = kk+16+hi*8+i
#pragma unroll
    for (int i = 0; i < 8; ++i) {
      bb[i]     = (_Float16)X[(size_t)(kk + hi * 8 + i) * N_ + n];
      bb[8 + i] = (_Float16)X[(size_t)(kk + 16 + hi * 8 + i) * N_ + n];
    }
    acc = wmma_f16(a, bb, acc);
  }

  // D layout: lane col = n (l2), row o = obase + r + hi*8
#pragma unroll
  for (int r = 0; r < 8; ++r) {
    const int o = obase + r + hi * 8;
    float y = acc[r] + bias[o];
    if (mat == 2) {
      vws[((size_t)b * C_ + o) * N_ + n] = (_Float16)y;          // [b, o, n]
    } else {
      const int h = o >> 5, d = o & 31;
      if (mat == 0) y *= SCALE_Q;
      _Float16* dst = (mat == 0) ? qws : kws;
      dst[((size_t)(b * NH_ + h) * N_ + n) * HD_ + d] = (_Float16)y;  // [b,h,n,d]
    }
  }
}

// =====================================================================
// Kernel 2: flash-attention.  grid = (N/64, NH, B), block = 128.
// Each wave owns 16 queries; streams keys 32 at a time.
//   s^T tile (keys x queries) D-layout == B-operand layout of p for the
//   attn*V WMMA -> softmax is lane-local except one shfl_xor(16).
// =====================================================================
__global__ __launch_bounds__(128) void attn_kernel(
    const _Float16* __restrict__ qws, const _Float16* __restrict__ kws,
    const _Float16* __restrict__ vws, const float* __restrict__ x,
    const float* __restrict__ gamma, float* __restrict__ out)
{
  const int lane = threadIdx.x & 31;
  const int wid  = threadIdx.x >> 5;
  const int l2   = lane & 15;
  const int hi   = lane >> 4;

  const int h  = blockIdx.y;
  const int b  = blockIdx.z;
  const int nq = (blockIdx.x * 4 + wid) * 16 + l2;   // this lane's query col

  // Q as resident B-operand (hd=32 K-dim x 16 queries)
  const _Float16* qp = qws + ((size_t)(b * NH_ + h) * N_ + nq) * HD_;
  const v16h qb = cat8(ld8(qp + hi * 8), ld8(qp + 16 + hi * 8));

  const _Float16* kbase = kws + (size_t)(b * NH_ + h) * N_ * HD_;
  const _Float16* vbase = vws + (size_t)(b * C_ + h * HD_) * N_;

  v8f acc0 = {0.f,0.f,0.f,0.f,0.f,0.f,0.f,0.f};
  v8f acc1 = {0.f,0.f,0.f,0.f,0.f,0.f,0.f,0.f};
  const v8f zero = {0.f,0.f,0.f,0.f,0.f,0.f,0.f,0.f};
  float m_run = -INFINITY, l_run = 0.0f;

  for (int j = 0; j < N_; j += 32) {
    // K^T tiles as A operands: (16 keys) x (32 hd), keys j..j+15 / j+16..j+31
    const _Float16* kp0 = kbase + (size_t)(j + l2) * HD_;
    const _Float16* kp1 = kbase + (size_t)(j + 16 + l2) * HD_;
    const v16h a0 = cat8(ld8(kp0 + hi * 8), ld8(kp0 + 16 + hi * 8));
    const v16h a1 = cat8(ld8(kp1 + hi * 8), ld8(kp1 + 16 + hi * 8));

    const v8f s0 = wmma_f16(a0, qb, zero);   // [key 0..15 x query]
    const v8f s1 = wmma_f16(a1, qb, zero);   // [key 16..31 x query]

    // --- online softmax over 32 keys for this lane's query ---
    float lm = -INFINITY;
#pragma unroll
    for (int r = 0; r < 8; ++r) lm = fmaxf(lm, fmaxf(s0[r], s1[r]));
    lm = fmaxf(lm, __shfl_xor(lm, 16, 32));  // combine lane pair (q, q+16)
    const float m_new = fmaxf(m_run, lm);
    const float alpha = __expf(m_run - m_new);
    m_run = m_new;

    v16h pb;
    float ls = 0.0f;
#pragma unroll
    for (int r = 0; r < 8; ++r) {
      const float p0 = __expf(s0[r] - m_new);
      const float p1 = __expf(s1[r] - m_new);
      ls += p0 + p1;
      pb[r]     = (_Float16)p0;   // halves 0..7  = keys j+hi*8+0..7
      pb[8 + r] = (_Float16)p1;   // halves 8..15 = keys j+16+hi*8+0..7
    }
    ls += __shfl_xor(ls, 16, 32);
    l_run = l_run * alpha + ls;

#pragma unroll
    for (int r = 0; r < 8; ++r) { acc0[r] *= alpha; acc1[r] *= alpha; }

    // V tiles as A operands: (16 d) x (32 keys)
    const _Float16* vp0 = vbase + (size_t)l2 * N_ + j;          // d = 0..15
    const _Float16* vp1 = vbase + (size_t)(16 + l2) * N_ + j;   // d = 16..31
    const v16h av0 = cat8(ld8(vp0 + hi * 8), ld8(vp0 + 16 + hi * 8));
    const v16h av1 = cat8(ld8(vp1 + hi * 8), ld8(vp1 + 16 + hi * 8));

    acc0 = wmma_f16(av0, pb, acc0);
    acc1 = wmma_f16(av1, pb, acc1);
  }

  const float g     = gamma[0];
  const float inv_l = 1.0f / l_run;
#pragma unroll
  for (int r = 0; r < 8; ++r) {
    const int d0 = r + hi * 8;
    const size_t row0 = (size_t)(b * C_ + h * HD_ + d0) * N_ + nq;
    const size_t row1 = row0 + (size_t)16 * N_;
    out[row0] = g * acc0[r] * inv_l + x[row0];
    out[row1] = g * acc1[r] * inv_l + x[row1];
  }
}

// =====================================================================
extern "C" void kernel_launch(void* const* d_in, const int* in_sizes, int n_in,
                              void* d_out, int out_size, void* d_ws, size_t ws_size,
                              hipStream_t stream) {
  const float* x  = (const float*)d_in[0];
  const float* wq = (const float*)d_in[1];
  const float* bq = (const float*)d_in[2];
  const float* wk = (const float*)d_in[3];
  const float* bk = (const float*)d_in[4];
  const float* wv = (const float*)d_in[5];
  const float* bv = (const float*)d_in[6];
  const float* gamma = (const float*)d_in[7];

  const int B = in_sizes[0] / (C_ * N_);   // = 4

  const size_t per = (size_t)B * C_ * N_;  // elements per f16 tensor
  _Float16* qws = (_Float16*)d_ws;
  _Float16* kws = qws + per;
  _Float16* vws = kws + per;

  dim3 g1(N_ / 16, 4, B * 3);              // (144, 4, 12)
  qkv_proj_kernel<<<g1, 128, 0, stream>>>(x, wq, bq, wk, bk, wv, bv,
                                          qws, kws, vws);

  dim3 g2(N_ / 64, NH_, B);                // (36, 8, 4)
  attn_kernel<<<g2, 128, 0, stream>>>(qws, kws, vws, x, gamma, (float*)d_out);
}
